// CausalSelfAttention_15453292331148
// MI455X (gfx1250) — matrix operand order repaired and checked by reading, not compiled, not run on hardware
//
#include <hip/hip_runtime.h>
#include <hip/hip_bf16.h>

typedef __bf16 bf16_t;
typedef __bf16 v16bf __attribute__((ext_vector_type(16)));
typedef __bf16 v8bf  __attribute__((ext_vector_type(8)));
typedef __bf16 v4bf  __attribute__((ext_vector_type(4)));
typedef float  v8f   __attribute__((ext_vector_type(8)));
typedef float  v4f   __attribute__((ext_vector_type(4)));
typedef unsigned int u32x4 __attribute__((ext_vector_type(4)));
typedef int          i32x4 __attribute__((ext_vector_type(4)));
typedef int          i32x8 __attribute__((ext_vector_type(8)));

#define WMMA_BF16(A, B, C) \
  __builtin_amdgcn_wmma_f32_16x16x32_bf16(false, (A), false, (B), (short)0, (C), false, false)

#if defined(__has_builtin)
#if __has_builtin(__builtin_amdgcn_tensor_load_to_lds)
#define HAVE_TDM 1
#endif
#endif

constexpr int kT = 2048;     // sequence length
constexpr int kC = 1024;     // embed dim
constexpr int kB = 4;        // batch
constexpr int kH = 16;       // heads
constexpr int kD = 64;       // head dim
constexpr int kM = kB * kT;  // 8192 rows

#if HAVE_TDM
// Issue a 2-D tensor DMA: global tile -> LDS (descriptor per ISA 08 §8).
__device__ __forceinline__ void tdm_load_2d(unsigned lds_off, const void* gptr, i32x8 g1) {
  const unsigned long long ga = (unsigned long long)(size_t)gptr;
  u32x4 g0 = {1u, lds_off, (unsigned)ga,
              ((unsigned)(ga >> 32) & 0x01FFFFFFu) | 0x80000000u};
  const i32x4 z4 = {0, 0, 0, 0};
#if __clang_major__ >= 23
  const i32x8 z8 = {0, 0, 0, 0, 0, 0, 0, 0};
  __builtin_amdgcn_tensor_load_to_lds(g0, g1, z4, z4, z8, 0);
#else
  __builtin_amdgcn_tensor_load_to_lds(g0, g1, z4, z4, 0);
#endif
}
__device__ __forceinline__ void tdm_wait0() {
#if __has_builtin(__builtin_amdgcn_s_wait_tensorcnt)
  __builtin_amdgcn_s_wait_tensorcnt(0);
#else
  asm volatile("s_wait_tensorcnt 0x0" ::: "memory");
#endif
}
#endif

// ---------------------------------------------------------------------------
// fp32 -> bf16 convert of activations (one pass, reused by 3 GEMMs)
// ---------------------------------------------------------------------------
__global__ __launch_bounds__(256) void x_convert_kernel(const float* __restrict__ x,
                                                        bf16_t* __restrict__ xb) {
  const int i = (blockIdx.x * 256 + threadIdx.x) * 4;
  v4f c = *(const v4f*)(x + i);
  v4bf o;
#pragma unroll
  for (int e = 0; e < 4; ++e) o[e] = (bf16_t)c[e];
  *(v4bf*)(xb + i) = o;
}

// ---------------------------------------------------------------------------
// Weight transpose + fp32->bf16 convert:  Wt[n][k] = (bf16) W[k][n]
// ---------------------------------------------------------------------------
__global__ __launch_bounds__(256) void wt_convert_kernel(const float* __restrict__ W,
                                                         bf16_t* __restrict__ Wt) {
  int idx = blockIdx.x * 256 + threadIdx.x;   // idx = n*1024 + k
  int n = idx >> 10;
  int k = idx & 1023;
  Wt[idx] = (bf16_t)W[k * kC + n];
}

// ---------------------------------------------------------------------------
// WMMA GEMM:  out = A(8192x1024,bf16) * Wt^T + bias.  Wt pre-transposed (N x K).
// Block: 256 threads = 8 waves; block tile 128x128; wave tile 32x64.
// A and W tiles (128 x 64 halves) staged in LDS by TDM, double-buffered,
// TDM pad: 32 dw data + 4 dw pad => 72-half row stride (36-bank, conflict-free).
// 16 WMMAs per barrier (two 32-deep sub-chunks).
// mode 0: Q->bf16 (B,H,T,D) (1/sqrt(D) folded); 1: K->(B,H,T,D);
// mode 2: V->bf16 (B,H,D,T); 3: fp32 (B,T,C).
// ---------------------------------------------------------------------------
constexpr int kGRow = 72;  // halves per padded LDS row (64 data + 8 pad)

__global__ __launch_bounds__(256) void gemm_wmma_kernel(const bf16_t* __restrict__ Axb,
                                                        const bf16_t* __restrict__ Wt,
                                                        const float* __restrict__ bias,
                                                        void* __restrict__ dst,
                                                        int mode, float scale) {
  __shared__ alignas(32) bf16_t As[2][128 * kGRow];  // 2 x 18 KB
  __shared__ alignas(32) bf16_t Ws[2][128 * kGRow];  // 2 x 18 KB

  const int lane = threadIdx.x & 31;
  const int wave = threadIdx.x >> 5;
  const int lo = lane & 15;
  const int hi = lane >> 4;
  const int wm = wave & 3;       // 4 waves along M
  const int wn = wave >> 2;      // 2 waves along N
  const int mBase = blockIdx.x * 128 + wm * 32;
  const int nBase = blockIdx.y * 128 + wn * 64;
  const int wave_s = __builtin_amdgcn_readfirstlane(wave);

  const bf16_t* Apanel = Axb + (size_t)(blockIdx.x * 128) * kC;
  const bf16_t* Wpanel = Wt + (size_t)(blockIdx.y * 128) * kC;

#if HAVE_TDM
  const unsigned ldsA[2] = {(unsigned)(size_t)&As[0][0], (unsigned)(size_t)&As[1][0]};
  const unsigned ldsW[2] = {(unsigned)(size_t)&Ws[0][0], (unsigned)(size_t)&Ws[1][0]};
  // data_size=2B | pad_enable | pad_interval=32dw | pad_amount=4dw; tile 64(k)x128
  const i32x8 g1W = {0x07110000, 0x04000000, 0x04000000, 0x00400000, 128, 1024, 0, 0};
  const i32x8 g1A = {0x07110000, 0x04000000, 0x20000000, 0x00400000, 128, 1024, 0, 0};
  if (wave_s == 0) {
    tdm_load_2d(ldsA[0], Apanel, g1A);
    tdm_load_2d(ldsW[0], Wpanel, g1W);
  }
#else
  {
    const int row = threadIdx.x >> 1, off = (threadIdx.x & 1) * 32;
    const bf16_t* sa = Apanel + (size_t)row * kC + off;
    const bf16_t* sw = Wpanel + (size_t)row * kC + off;
    bf16_t* da = &As[0][row * kGRow + off];
    bf16_t* dw = &Ws[0][row * kGRow + off];
#pragma unroll
    for (int q = 0; q < 4; ++q) {
      *(v8bf*)(da + q * 8) = *(const v8bf*)(sa + q * 8);
      *(v8bf*)(dw + q * 8) = *(const v8bf*)(sw + q * 8);
    }
  }
#endif

  v8f acc[2][4];
#pragma unroll
  for (int mt = 0; mt < 2; ++mt)
#pragma unroll
    for (int nt = 0; nt < 4; ++nt)
#pragma unroll
      for (int r = 0; r < 8; ++r) acc[mt][nt][r] = 0.0f;

  int ib = 0;
  for (int kc = 0; kc < kC; kc += 64, ib ^= 1) {
#if HAVE_TDM
    if (wave_s == 0) tdm_wait0();  // tiles for buffer ib complete
#endif
    __syncthreads();

    // stage next K-chunk into the other buffer (overlaps with compute below)
    if (kc + 64 < kC) {
#if HAVE_TDM
      if (wave_s == 0) {
        tdm_load_2d(ldsA[ib ^ 1], Apanel + kc + 64, g1A);
        tdm_load_2d(ldsW[ib ^ 1], Wpanel + kc + 64, g1W);
      }
#else
      const int row = threadIdx.x >> 1, off = (threadIdx.x & 1) * 32;
      const bf16_t* sa = Apanel + (size_t)row * kC + kc + 64 + off;
      const bf16_t* sw = Wpanel + (size_t)row * kC + kc + 64 + off;
      bf16_t* da = &As[ib ^ 1][row * kGRow + off];
      bf16_t* dw = &Ws[ib ^ 1][row * kGRow + off];
#pragma unroll
      for (int q = 0; q < 4; ++q) {
        *(v8bf*)(da + q * 8) = *(const v8bf*)(sa + q * 8);
        *(v8bf*)(dw + q * 8) = *(const v8bf*)(sw + q * 8);
      }
#endif
    }

    // two 32-deep sub-chunks => 16 WMMAs between barriers
#pragma unroll
    for (int ks = 0; ks < 64; ks += 32) {
      // ---- A fragments from LDS: lane m-row = wm*32+mt*16+lo
      v16bf afrag[2];
#pragma unroll
      for (int mt = 0; mt < 2; ++mt) {
        const bf16_t* p = &As[ib][(wm * 32 + mt * 16 + lo) * kGRow + ks + hi * 8];
        v8bf c1 = *(const v8bf*)p;
        v8bf c2 = *(const v8bf*)(p + 16);
#pragma unroll
        for (int e = 0; e < 8; ++e) {
          afrag[mt][e]     = c1[e];
          afrag[mt][e + 8] = c2[e];
        }
      }
      // ---- B fragments from LDS: lane n-row = wn*64+nt*16+lo; k = ks+8*hi+e, ks+16+8*hi+e
      v16bf bfrag[4];
#pragma unroll
      for (int nt = 0; nt < 4; ++nt) {
        const bf16_t* p = &Ws[ib][(wn * 64 + nt * 16 + lo) * kGRow + ks + hi * 8];
        v8bf c1 = *(const v8bf*)p;
        v8bf c2 = *(const v8bf*)(p + 16);
#pragma unroll
        for (int e = 0; e < 8; ++e) {
          bfrag[nt][e]     = c1[e];
          bfrag[nt][e + 8] = c2[e];
        }
      }
#pragma unroll
      for (int mt = 0; mt < 2; ++mt)
#pragma unroll
        for (int nt = 0; nt < 4; ++nt)
          acc[mt][nt] = WMMA_BF16(afrag[mt], bfrag[nt], acc[mt][nt]);
    }
  }

  // ---- epilogue: C layout lane holds (m = r + 8*hi, n = lo)
#pragma unroll
  for (int nt = 0; nt < 4; ++nt) {
    const int col = nBase + nt * 16 + lo;
    const float bv = bias[col];
    const int h = col >> 6;
    const int d = col & 63;
#pragma unroll
    for (int mt = 0; mt < 2; ++mt) {
#pragma unroll
      for (int r = 0; r < 8; ++r) {
        const int row = mBase + mt * 16 + r + hi * 8;
        const float v = (acc[mt][nt][r] + bv) * scale;
        const int b = row >> 11;
        const int t = row & 2047;
        if (mode == 0 || mode == 1) {
          ((bf16_t*)dst)[(((size_t)(b * kH + h) * kT) + t) * kD + d] = (bf16_t)v;
        } else if (mode == 2) {
          ((bf16_t*)dst)[((size_t)(b * kH + h) * kD + d) * kT + t] = (bf16_t)v;
        } else {
          ((float*)dst)[(size_t)row * kC + col] = v;
        }
      }
    }
  }
}

// ---------------------------------------------------------------------------
// Flash-style causal attention. Grid (B*H, T/64); 4 waves per block.
// 64-key K/V tiles double-buffered in LDS via TDM (padded rows, conflict-free),
// processed as two 32-key subtiles with scalar diagonal/interior paths.
// Row sums via one extra WMMA (P x ones).
// Q,K: bf16 (B,H,T,D).  Vt: bf16 (B,H,D,T).  AO: bf16 (B,T,C).
// ---------------------------------------------------------------------------
#define ATT_WAVES 4
constexpr int kARow = 72;  // padded LDS row for K/V tiles (64 data + 8 pad)
constexpr int kPRow = 40;  // padded LDS row for P scratch (32 data + 8 pad)

__global__ __launch_bounds__(32 * ATT_WAVES) void attn_kernel(const bf16_t* __restrict__ Q,
                                                              const bf16_t* __restrict__ K,
                                                              const bf16_t* __restrict__ Vt,
                                                              bf16_t* __restrict__ AO) {
  __shared__ alignas(32) bf16_t Ks[2][64 * kARow];         // 2 x 9 KB (keys x d)
  __shared__ alignas(32) bf16_t Vs[2][64 * kARow];         // 2 x 9 KB (d x keys)
  __shared__ alignas(32) bf16_t Pl[ATT_WAVES][16 * kPRow]; // per-wave P tile

  const int tid = threadIdx.x;
  const int lane = tid & 31;
  const int wave = tid >> 5;
  const int lo = lane & 15;
  const int hi = lane >> 4;
  const int bh = blockIdx.x;                                  // b*16 + h
  const int qbase = (blockIdx.y * ATT_WAVES + wave) * 16;
  const int kmax_blk = (blockIdx.y + 1) * (ATT_WAVES * 16);
  const int wave_s = __builtin_amdgcn_readfirstlane(wave);
  const int qbase_s = __builtin_amdgcn_readfirstlane(qbase);  // wave-uniform
  const int kmaxw_s = qbase_s + 16;

  const bf16_t* Qh = Q + (size_t)bh * kT * kD;
  const bf16_t* Kh = K + (size_t)bh * kT * kD;
  const bf16_t* Vh = Vt + (size_t)bh * kD * kT;

#if HAVE_TDM
  const unsigned ldsK[2] = {(unsigned)(size_t)&Ks[0][0], (unsigned)(size_t)&Ks[1][0]};
  const unsigned ldsV[2] = {(unsigned)(size_t)&Vs[0][0], (unsigned)(size_t)&Vs[1][0]};
  // pad: 32 dw data + 4 dw => 72-half rows.
  // K: tensor 64x2048 stride 64, tile 64(d)x64(keys)
  // V: tensor 2048x64 stride 2048, tile 64(t)x64(d)
  const i32x8 g1K = {0x07110000, 0x00400000, 0x08000000, 0x00400000, 64, 64, 0, 0};
  const i32x8 g1V = {0x07110000, 0x08000000, 0x00400000, 0x00400000, 64, 2048, 0, 0};
  if (wave_s == 0) {
    tdm_load_2d(ldsK[0], Kh, g1K);
    tdm_load_2d(ldsV[0], Vh, g1V);
  }
#else
  {
    const int row = tid >> 1, off = (tid & 1) * 32;
    bf16_t* dk = &Ks[0][row * kARow + off];
    bf16_t* dv = &Vs[0][row * kARow + off];
    const bf16_t* sk = Kh + (size_t)row * kD + off;
    const bf16_t* sv = Vh + (size_t)row * kT + off;
#pragma unroll
    for (int q = 0; q < 4; ++q) {
      *(v8bf*)(dk + q * 8) = *(const v8bf*)(sk + q * 8);
      *(v8bf*)(dv + q * 8) = *(const v8bf*)(sv + q * 8);
    }
  }
#endif

  // Q fragments for d-chunks [0,32) and [32,64)
  v16bf aq[2];
#pragma unroll
  for (int dc = 0; dc < 2; ++dc) {
    const bf16_t* p = Qh + (size_t)(qbase + lo) * kD + dc * 32 + hi * 16;
    v8bf c1 = *(const v8bf*)p;
    v8bf c2 = *(const v8bf*)(p + 8);
#pragma unroll
    for (int e = 0; e < 8; ++e) {
      aq[dc][e]     = c1[e];
      aq[dc][e + 8] = c2[e];
    }
  }

  v16bf ones;
#pragma unroll
  for (int e = 0; e < 16; ++e) ones[e] = (bf16_t)1.0f;

  const float NEG_INF = -__builtin_inff();
  float rmax[8], rsum[8];
  v8f o[4];
#pragma unroll
  for (int r = 0; r < 8; ++r) { rmax[r] = NEG_INF; rsum[r] = 0.0f; }
#pragma unroll
  for (int dt = 0; dt < 4; ++dt)
#pragma unroll
    for (int r = 0; r < 8; ++r) o[dt][r] = 0.0f;

  int ib = 0;
  for (int kb = 0; kb < kmax_blk; kb += 64, ib ^= 1) {
#if HAVE_TDM
    if (wave_s == 0) tdm_wait0();  // buffer ib ready
#endif
    __syncthreads();

    // stage the next 64-key K/V tile into the other buffer (overlaps compute)
    if (kb + 64 < kmax_blk) {
#if HAVE_TDM
      if (wave_s == 0) {
        tdm_load_2d(ldsK[ib ^ 1], Kh + (size_t)(kb + 64) * kD, g1K);
        tdm_load_2d(ldsV[ib ^ 1], Vh + kb + 64, g1V);
      }
#else
      const int row = tid >> 1, off = (tid & 1) * 32;
      bf16_t* dk = &Ks[ib ^ 1][row * kARow + off];
      bf16_t* dv = &Vs[ib ^ 1][row * kARow + off];
      const bf16_t* sk = Kh + (size_t)(kb + 64 + row) * kD + off;
      const bf16_t* sv = Vh + (size_t)row * kT + kb + 64 + off;
#pragma unroll
      for (int q = 0; q < 4; ++q) {
        *(v8bf*)(dk + q * 8) = *(const v8bf*)(sk + q * 8);
        *(v8bf*)(dv + q * 8) = *(const v8bf*)(sv + q * 8);
      }
#endif
    }

    // One 32-key subtile: S wmmas, online softmax, PV wmmas.  domask is
    // constant at each call site so mask code exists only on diagonal path.
    auto compute_tile = [&](const int ksub, const bool domask) {
      const int kbs = kb + ksub;
      v8f s0, s1;
#pragma unroll
      for (int r = 0; r < 8; ++r) { s0[r] = 0.0f; s1[r] = 0.0f; }
#pragma unroll
      for (int dc = 0; dc < 2; ++dc) {
        v16bf b0 = *(const v16bf*)(&Ks[ib][(ksub + lo) * kARow + dc * 32 + hi * 16]);
        v16bf b1 = *(const v16bf*)(&Ks[ib][(ksub + 16 + lo) * kARow + dc * 32 + hi * 16]);
        s0 = WMMA_BF16(aq[dc], b0, s0);
        s1 = WMMA_BF16(aq[dc], b1, s1);
      }

      float al[8];
#pragma unroll
      for (int r = 0; r < 8; ++r) {
        float x0 = s0[r];
        float x1 = s1[r];
        if (domask) {
          const int query = qbase + r + hi * 8;
          x0 = ((kbs + lo) > query) ? NEG_INF : x0;
          x1 = ((kbs + 16 + lo) > query) ? NEG_INF : x1;
        }
        float v = fmaxf(x0, x1);
        v = fmaxf(v, __shfl_xor(v, 1, 32));
        v = fmaxf(v, __shfl_xor(v, 2, 32));
        v = fmaxf(v, __shfl_xor(v, 4, 32));
        v = fmaxf(v, __shfl_xor(v, 8, 32));
        const float nm = fmaxf(rmax[r], v);
        al[r] = __expf(rmax[r] - nm);
        rmax[r] = nm;
        const float p0 = __expf(x0 - nm);
        const float p1 = __expf(x1 - nm);
        o[0][r] *= al[r];
        o[1][r] *= al[r];
        o[2][r] *= al[r];
        o[3][r] *= al[r];
        Pl[wave][(r + 8 * hi) * kPRow + lo]      = (bf16_t)p0;
        Pl[wave][(r + 8 * hi) * kPRow + 16 + lo] = (bf16_t)p1;
      }
      // Same-wave LDS ops are in-order; just stop compiler reordering.
      asm volatile("" ::: "memory");

      // ---- repack P (C layout) -> A layout via per-wave LDS region
      v16bf ap;
      {
        v8bf c1 = *(const v8bf*)(&Pl[wave][lo * kPRow + hi * 16]);
        v8bf c2 = *(const v8bf*)(&Pl[wave][lo * kPRow + hi * 16 + 8]);
#pragma unroll
        for (int e = 0; e < 8; ++e) {
          ap[e]     = c1[e];
          ap[e + 8] = c2[e];
        }
      }

      // ---- row sums with one WMMA: (P x ones)[m][*] = sum_k P[m][k]
      v8f zero8;
#pragma unroll
      for (int r = 0; r < 8; ++r) zero8[r] = 0.0f;
      v8f sumt = WMMA_BF16(ap, ones, zero8);
#pragma unroll
      for (int r = 0; r < 8; ++r) rsum[r] = rsum[r] * al[r] + sumt[r];

      // ---- O += P * V  (K-dim = 32 keys, 4 N-tiles across D=64)
#pragma unroll
      for (int dt = 0; dt < 4; ++dt) {
        v16bf bv = *(const v16bf*)(&Vs[ib][(dt * 16 + lo) * kARow + ksub + hi * 16]);
        o[dt] = WMMA_BF16(ap, bv, o[dt]);
      }
    };

#pragma unroll
    for (int ksub = 0; ksub < 64; ksub += 32) {
      const int kbs = kb + ksub;
      if (kbs < kmaxw_s) {            // scalar causal activity test
        if (kbs + 31 > qbase_s) {
          compute_tile(ksub, true);   // diagonal subtile: apply causal mask
        } else {
          compute_tile(ksub, false);  // interior subtile: no masking
        }
      }
    }
  }

  // ---- normalize and store to (B, T, C) as bf16
  const int b = bh >> 4;
  const int h = bh & 15;
#pragma unroll
  for (int dt = 0; dt < 4; ++dt) {
#pragma unroll
    for (int r = 0; r < 8; ++r) {
      const float val = o[dt][r] / rsum[r];
      const size_t row = (size_t)(b * kT + qbase + r + hi * 8);
      AO[row * kC + h * kD + dt * 16 + lo] = (bf16_t)val;
    }
  }
}

// ---------------------------------------------------------------------------
// Launch
// ---------------------------------------------------------------------------
extern "C" void kernel_launch(void* const* d_in, const int* in_sizes, int n_in,
                              void* d_out, int out_size, void* d_ws, size_t ws_size,
                              hipStream_t stream) {
  (void)in_sizes; (void)n_in; (void)out_size; (void)ws_size;
  const float* x  = (const float*)d_in[0];
  const float* Wq = (const float*)d_in[1];
  const float* bq = (const float*)d_in[2];
  const float* Wk = (const float*)d_in[3];
  const float* bk = (const float*)d_in[4];
  const float* Wv = (const float*)d_in[5];
  const float* bv = (const float*)d_in[6];
  const float* Wp = (const float*)d_in[7];
  const float* bp = (const float*)d_in[8];

  char* ws = (char*)d_ws;
  const size_t MB = 1024ull * 1024ull;
  bf16_t* Wqt = (bf16_t*)(ws + 0 * MB);   // 2 MB each
  bf16_t* Wkt = (bf16_t*)(ws + 2 * MB);
  bf16_t* Wvt = (bf16_t*)(ws + 4 * MB);
  bf16_t* Wpt = (bf16_t*)(ws + 6 * MB);
  bf16_t* xb  = (bf16_t*)(ws + 8 * MB);   // 16 MB bf16 activations
  bf16_t* Qb  = (bf16_t*)(ws + 24 * MB);  // 16 MB each
  bf16_t* Kb  = (bf16_t*)(ws + 40 * MB);
  bf16_t* Vtb = (bf16_t*)(ws + 56 * MB);
  bf16_t* AOb = (bf16_t*)(ws + 72 * MB);  // total 88 MB

  x_convert_kernel<<<(kM * kC) / (256 * 4), 256, 0, stream>>>(x, xb);
  const int wtBlocks = (kC * kC) / 256;
  wt_convert_kernel<<<wtBlocks, 256, 0, stream>>>(Wq, Wqt);
  wt_convert_kernel<<<wtBlocks, 256, 0, stream>>>(Wk, Wkt);
  wt_convert_kernel<<<wtBlocks, 256, 0, stream>>>(Wv, Wvt);
  wt_convert_kernel<<<wtBlocks, 256, 0, stream>>>(Wp, Wpt);

  dim3 ggrid(kM / 128, kC / 128);
  const float qscale = 0.125f;  // 1/sqrt(64)
  gemm_wmma_kernel<<<ggrid, 256, 0, stream>>>(xb, Wqt, bq, Qb, 0, qscale);
  gemm_wmma_kernel<<<ggrid, 256, 0, stream>>>(xb, Wkt, bk, Kb, 1, 1.0f);
  gemm_wmma_kernel<<<ggrid, 256, 0, stream>>>(xb, Wvt, bv, Vtb, 2, 1.0f);

  attn_kernel<<<dim3(kB * kH, kT / (16 * ATT_WAVES)), 32 * ATT_WAVES, 0, stream>>>(
      Qb, Kb, Vtb, AOb);

  gemm_wmma_kernel<<<ggrid, 256, 0, stream>>>(AOb, Wpt, bp, d_out, 3, 1.0f);
}
